// e2e_58025008169157
// MI455X (gfx1250) — matrix operand order repaired and checked by reading, not compiled, not run on hardware
//
#include <hip/hip_runtime.h>
#include <hip/hip_bf16.h>

// ---------------------------------------------------------------------------
// CDNA5 (gfx1250) implementation.
// GEMMs run on v_wmma_f32_16x16x32_bf16 with a bf16 hi/lo compensated split
// (3 WMMAs per K=32 step -> ~fp32 accuracy at bf16-matrix throughput).
// The fp32->bf16(hi,lo) split is done ONCE in a streaming pass; the GEMM
// inner loop is pure b128 loads + WMMA. Wave->tile mapping makes the 8 waves
// of a workgroup share the SAME B columns (2/3 of traffic -> WGP$ hits).
// ---------------------------------------------------------------------------

typedef __attribute__((ext_vector_type(16))) __bf16 v16bf;
typedef __attribute__((ext_vector_type(8)))  float  v8f;

union BFrag { v16bf v; uint4 q[2]; };   // q[0] = elems 0..7, q[1] = elems 8..15

__device__ __forceinline__ v8f wmma_bf3(const BFrag& ah, const BFrag& al,
                                        const BFrag& bh, const BFrag& bl, v8f acc) {
    acc = __builtin_amdgcn_wmma_f32_16x16x32_bf16(false, ah.v, false, bh.v, (short)0, acc, false, false);
    acc = __builtin_amdgcn_wmma_f32_16x16x32_bf16(false, al.v, false, bh.v, (short)0, acc, false, false);
    acc = __builtin_amdgcn_wmma_f32_16x16x32_bf16(false, ah.v, false, bl.v, (short)0, acc, false, false);
    return acc;
}

// ---------------------------------------------------------------------------
// Streaming split: fp32 -> bf16 hi + bf16 lo residual (RNE both stages).
// ---------------------------------------------------------------------------
__global__ __launch_bounds__(256) void k_split(const float* __restrict__ in,
                                               __bf16* __restrict__ hi,
                                               __bf16* __restrict__ lo, size_t n) {
    size_t gid = (size_t)blockIdx.x * blockDim.x + threadIdx.x;
    size_t n4 = n >> 2;
    if (gid < n4) {
        float4 x = reinterpret_cast<const float4*>(in)[gid];
        union { __bf16 h[4]; uint2 u; } H, L;
        float v[4] = {x.x, x.y, x.z, x.w};
#pragma unroll
        for (int e = 0; e < 4; ++e) {
            __bf16 h = (__bf16)v[e];
            H.h[e] = h;
            L.h[e] = (__bf16)(v[e] - (float)h);
        }
        reinterpret_cast<uint2*>(hi)[gid] = H.u;
        reinterpret_cast<uint2*>(lo)[gid] = L.u;
    }
    if (gid == 0) {                      // scalar tail (n % 4)
        for (size_t r = n4 << 2; r < n; ++r) {
            __bf16 h = (__bf16)in[r];
            hi[r] = h;
            lo[r] = (__bf16)(in[r] - (float)h);
        }
    }
}

// ---------------------------------------------------------------------------
// GEMM1: Y_hat = X @ W^T + b. One wave -> 32x64 output block (2x4 subtiles).
// Writes epT[j][t] = Y[t][j] - Y_hat[t][j] for t < n_obs; yhat[j] at t==n_obs.
// Waves within a workgroup share B (same tN4 group) for WGP$ reuse.
// ---------------------------------------------------------------------------
__global__ __launch_bounds__(256) void k_gemm1(
    const __bf16* __restrict__ Xh, const __bf16* __restrict__ Xl,
    const __bf16* __restrict__ Wh, const __bf16* __restrict__ Wl,
    const float* __restrict__ bvec, const float* __restrict__ Y,
    float* __restrict__ epT, float* __restrict__ yhat,
    int rows, int n_x, int n_y, int n_obs, int tilesM2, int tilesN4) {
    int wave = (blockIdx.x * blockDim.x + threadIdx.x) >> 5;
    int lane = threadIdx.x & 31;
    if (wave >= tilesM2 * tilesN4) return;
    int tM2 = wave % tilesM2, tN4 = wave / tilesM2;   // block-mates share tN4
    int half = lane >> 4, l15 = lane & 15;

    int r0 = tM2 * 32 + l15;      if (r0 > rows - 1) r0 = rows - 1;
    int r1 = tM2 * 32 + 16 + l15; if (r1 > rows - 1) r1 = rows - 1;
    const __bf16* a0h = Xh + (size_t)r0 * n_x;
    const __bf16* a0l = Xl + (size_t)r0 * n_x;
    const __bf16* a1h = Xh + (size_t)r1 * n_x;
    const __bf16* a1l = Xl + (size_t)r1 * n_x;

    int cols[4]; const __bf16* bh[4]; const __bf16* bl[4];
#pragma unroll
    for (int s = 0; s < 4; ++s) {
        int cb = (tN4 * 4 + s) * 16 + l15;
        cols[s] = cb;
        int rb = cb > n_y - 1 ? n_y - 1 : cb;
        bh[s] = Wh + (size_t)rb * n_x;
        bl[s] = Wl + (size_t)rb * n_x;
    }

    const int o0 = half * 8, o1 = 16 + half * 8, ob = half * 16;
    v8f acc[2][4] = {};
    for (int k0 = 0; k0 < n_x; k0 += 32) {
        BFrag ah0, al0, ah1, al1;
        ah0.q[0] = *(const uint4*)(a0h + k0 + o0); ah0.q[1] = *(const uint4*)(a0h + k0 + o1);
        al0.q[0] = *(const uint4*)(a0l + k0 + o0); al0.q[1] = *(const uint4*)(a0l + k0 + o1);
        ah1.q[0] = *(const uint4*)(a1h + k0 + o0); ah1.q[1] = *(const uint4*)(a1h + k0 + o1);
        al1.q[0] = *(const uint4*)(a1l + k0 + o0); al1.q[1] = *(const uint4*)(a1l + k0 + o1);
#pragma unroll
        for (int s = 0; s < 4; ++s) {
            BFrag fbh, fbl;
            const __bf16* pb = bh[s] + k0 + o0;
            const __bf16* pl = bl[s] + k0 + o0;
            fbh.q[0] = *(const uint4*)pb; fbh.q[1] = *(const uint4*)(pb + 16);
            fbl.q[0] = *(const uint4*)pl; fbl.q[1] = *(const uint4*)(pl + 16);
            acc[0][s] = wmma_bf3(ah0, al0, fbh, fbl, acc[0][s]);
            acc[1][s] = wmma_bf3(ah1, al1, fbh, fbl, acc[1][s]);
        }
    }

#pragma unroll
    for (int m = 0; m < 2; ++m) {
#pragma unroll
        for (int s = 0; s < 4; ++s) {
            int col = cols[s];
            if (col < n_y) {
                float bc = bvec[col];
#pragma unroll
                for (int v = 0; v < 8; ++v) {
                    int t = tM2 * 32 + m * 16 + v + 8 * half;
                    float val = acc[m][s][v] + bc;
                    if (t < n_obs)
                        epT[(size_t)col * n_obs + t] = Y[(size_t)t * n_y + col] - val;
                    else if (t == n_obs)
                        yhat[col] = val;
                }
            }
        }
    }
}

// ---------------------------------------------------------------------------
// GEMM2: Cov = epT @ epT^T / n_obs - mu mu^T  (epT hi/lo are n_y x n_obs)
// ---------------------------------------------------------------------------
__global__ __launch_bounds__(256) void k_gemm2(
    const __bf16* __restrict__ Eh, const __bf16* __restrict__ El,
    const float* __restrict__ mu, float* __restrict__ Cov,
    int n_y, int n_obs, int tilesM2, int tilesN4) {
    int wave = (blockIdx.x * blockDim.x + threadIdx.x) >> 5;
    int lane = threadIdx.x & 31;
    if (wave >= tilesM2 * tilesN4) return;
    int tM2 = wave % tilesM2, tN4 = wave / tilesM2;   // block-mates share tN4
    int half = lane >> 4, l15 = lane & 15;

    int r0 = tM2 * 32 + l15;      if (r0 > n_y - 1) r0 = n_y - 1;
    int r1 = tM2 * 32 + 16 + l15; if (r1 > n_y - 1) r1 = n_y - 1;
    const __bf16* a0h = Eh + (size_t)r0 * n_obs;
    const __bf16* a0l = El + (size_t)r0 * n_obs;
    const __bf16* a1h = Eh + (size_t)r1 * n_obs;
    const __bf16* a1l = El + (size_t)r1 * n_obs;

    int cols[4]; const __bf16* bh[4]; const __bf16* bl[4];
#pragma unroll
    for (int s = 0; s < 4; ++s) {
        int cb = (tN4 * 4 + s) * 16 + l15;
        cols[s] = cb;
        int rb = cb > n_y - 1 ? n_y - 1 : cb;
        bh[s] = Eh + (size_t)rb * n_obs;
        bl[s] = El + (size_t)rb * n_obs;
    }

    const int o0 = half * 8, o1 = 16 + half * 8, ob = half * 16;
    v8f acc[2][4] = {};
    for (int k0 = 0; k0 < n_obs; k0 += 32) {
        BFrag ah0, al0, ah1, al1;
        ah0.q[0] = *(const uint4*)(a0h + k0 + o0); ah0.q[1] = *(const uint4*)(a0h + k0 + o1);
        al0.q[0] = *(const uint4*)(a0l + k0 + o0); al0.q[1] = *(const uint4*)(a0l + k0 + o1);
        ah1.q[0] = *(const uint4*)(a1h + k0 + o0); ah1.q[1] = *(const uint4*)(a1h + k0 + o1);
        al1.q[0] = *(const uint4*)(a1l + k0 + o0); al1.q[1] = *(const uint4*)(a1l + k0 + o1);
#pragma unroll
        for (int s = 0; s < 4; ++s) {
            BFrag fbh, fbl;
            const __bf16* pb = bh[s] + k0 + o0;
            const __bf16* pl = bl[s] + k0 + o0;
            fbh.q[0] = *(const uint4*)pb; fbh.q[1] = *(const uint4*)(pb + 16);
            fbl.q[0] = *(const uint4*)pl; fbl.q[1] = *(const uint4*)(pl + 16);
            acc[0][s] = wmma_bf3(ah0, al0, fbh, fbl, acc[0][s]);
            acc[1][s] = wmma_bf3(ah1, al1, fbh, fbl, acc[1][s]);
        }
    }

    float inv = 1.0f / (float)n_obs;
#pragma unroll
    for (int m = 0; m < 2; ++m) {
#pragma unroll
        for (int s = 0; s < 4; ++s) {
            int col = cols[s];
            if (col < n_y) {
                float muc = mu[col];
#pragma unroll
                for (int v = 0; v < 8; ++v) {
                    int r = tM2 * 32 + m * 16 + v + 8 * half;
                    if (r < n_y)
                        Cov[(size_t)r * n_y + col] = acc[m][s][v] * inv - mu[r] * muc;
                }
            }
        }
    }
}

// ---------------------------------------------------------------------------
// Helpers: wave reduce, row means, matvec
// ---------------------------------------------------------------------------
__device__ __forceinline__ float wave_sum(float s) {
#pragma unroll
    for (int m = 16; m; m >>= 1) s += __shfl_xor(s, m, 32);
    return s;
}

__global__ __launch_bounds__(256) void k_rowmean(
    const float* __restrict__ epT, float* __restrict__ mu, int n_obs, int n_y) {
    int wave = (blockIdx.x * blockDim.x + threadIdx.x) >> 5;
    int lane = threadIdx.x & 31;
    if (wave >= n_y) return;
    const float* row = epT + (size_t)wave * n_obs;
    float s = 0.f;
    if ((n_obs & 127) == 0) {
        for (int c = lane * 4; c < n_obs; c += 128) {
            float4 f = *reinterpret_cast<const float4*>(row + c);
            s += f.x + f.y + f.z + f.w;
        }
    } else {
        for (int c = lane; c < n_obs; c += 32) s += row[c];
    }
    s = wave_sum(s);
    if (lane == 0) mu[wave] = s / (float)n_obs;
}

__global__ __launch_bounds__(256) void k_matvec(
    const float* __restrict__ M, const float* __restrict__ x,
    float* __restrict__ out, int n) {
    int wave = (blockIdx.x * blockDim.x + threadIdx.x) >> 5;
    int lane = threadIdx.x & 31;
    if (wave >= n) return;
    const float* row = M + (size_t)wave * n;
    float s = 0.f;
    if ((n & 127) == 0) {
        for (int c = lane * 4; c < n; c += 128) {
            float4 f = *reinterpret_cast<const float4*>(row + c);
            float4 g = *reinterpret_cast<const float4*>(x + c);
            s += f.x * g.x + f.y * g.y + f.z * g.z + f.w * g.w;
        }
    } else {
        for (int c = lane; c < n; c += 32) s += row[c] * x[c];
    }
    s = wave_sum(s);
    if (lane == 0) out[wave] = s;
}

// ---------------------------------------------------------------------------
// Solver scalar-vector kernels
// ---------------------------------------------------------------------------
__global__ void k_init(const float* __restrict__ gamma, const float* __restrict__ yhat,
                       float* __restrict__ lin, float* __restrict__ z, float* __restrict__ y,
                       float* __restrict__ v, float* __restrict__ tL, int n) {
    int i = blockIdx.x * blockDim.x + threadIdx.x;
    if (i < n) {
        lin[i] = gamma[0] * yhat[i];
        float invn = 1.f / (float)n;
        z[i] = invn;
        y[i] = invn;
        v[i] = 1.f / sqrtf((float)n);
    }
    if (i == 0) tL[0] = 1.f;   // t
}

__global__ __launch_bounds__(1024) void k_normalize(
    const float* __restrict__ w, float* __restrict__ v, int n) {
    __shared__ float red[1024];
    int tid = threadIdx.x, nt = blockDim.x;
    float s = 0.f;
    for (int i = tid; i < n; i += nt) { float a = w[i]; s += a * a; }
    red[tid] = s; __syncthreads();
    for (int m = nt >> 1; m; m >>= 1) {
        if (tid < m) red[tid] += red[tid + m];
        __syncthreads();
    }
    float nrm = sqrtf(red[0]) + 1e-12f;
    for (int i = tid; i < n; i += nt) v[i] = w[i] / nrm;
}

__global__ __launch_bounds__(1024) void k_L(
    const float* __restrict__ v, const float* __restrict__ w,
    float* __restrict__ tL, int n) {
    __shared__ float red[1024];
    int tid = threadIdx.x, nt = blockDim.x;
    float s = 0.f;
    for (int i = tid; i < n; i += nt) s += v[i] * w[i];
    red[tid] = s; __syncthreads();
    for (int m = nt >> 1; m; m >>= 1) {
        if (tid < m) red[tid] += red[tid + m];
        __syncthreads();
    }
    if (tid == 0) tL[1] = 2.f * red[0] + 1e-6f;  // L
}

// ---------------------------------------------------------------------------
// One APG step: v = y - g/L, z = proj_simplex(v) via bitonic sort + scan,
// Nesterov momentum update. Single workgroup (1024 threads).
// ---------------------------------------------------------------------------
__global__ __launch_bounds__(1024) void k_apg(
    const float* __restrict__ w, const float* __restrict__ lin,
    float* __restrict__ z, float* __restrict__ y,
    float* __restrict__ tL, int n, int np2) {
    extern __shared__ float smem[];
    float* u  = smem;              // np2 : sorted values
    float* sa = smem + np2;        // np2 : scan buffer A
    float* sb = smem + 2 * np2;    // np2 : scan buffer B
    __shared__ int   rho_s;
    __shared__ float theta_s, tnew_s;

    int tid = threadIdx.x, nt = blockDim.x;
    float L = tL[1], t = tL[0];

    float vreg[4];
    int c = 0;
    for (int i = tid; i < np2; i += nt, ++c) {
        float vi = (i < n) ? (y[i] - (2.f * w[i] - lin[i]) / L) : -__builtin_inff();
        u[i] = vi;
        if (c < 4) vreg[c] = vi;
    }
    if (tid == 0) rho_s = 0;
    __syncthreads();

    // Bitonic sort, descending.
    for (int k = 2; k <= np2; k <<= 1) {
        for (int j = k >> 1; j > 0; j >>= 1) {
            for (int i = tid; i < np2; i += nt) {
                int ixj = i ^ j;
                if (ixj > i) {
                    float a = u[i], d = u[ixj];
                    bool sw = ((i & k) == 0) ? (a < d) : (a > d);
                    if (sw) { u[i] = d; u[ixj] = a; }
                }
            }
            __syncthreads();
        }
    }

    // Inclusive scan (Hillis-Steele, double buffered).
    for (int i = tid; i < np2; i += nt) sa[i] = u[i];
    __syncthreads();
    float* src = sa; float* dst = sb;
    for (int off = 1; off < np2; off <<= 1) {
        for (int i = tid; i < np2; i += nt)
            dst[i] = src[i] + ((i >= off) ? src[i - off] : 0.f);
        __syncthreads();
        float* tmp = src; src = dst; dst = tmp;
    }

    // rho = #{i : u_i*(i+1) > cumsum_i - 1}
    int cnt = 0;
    for (int i = tid; i < n; i += nt) {
        float css = src[i] - 1.f;
        if (u[i] * (float)(i + 1) > css) cnt++;
    }
    atomicAdd(&rho_s, cnt);
    __syncthreads();
    if (tid == 0) {
        int rho = rho_s > 0 ? rho_s : 1;
        theta_s = (src[rho - 1] - 1.f) / (float)rho;
        tnew_s  = 0.5f * (1.f + sqrtf(1.f + 4.f * t * t));
    }
    __syncthreads();

    float theta = theta_s, t_new = tnew_s;
    float coef = (t - 1.f) / t_new;
    c = 0;
    for (int i = tid; i < n; i += nt, ++c) {
        float zn = fmaxf(vreg[c] - theta, 0.f);
        float yn = zn + coef * (zn - z[i]);
        z[i] = zn;
        y[i] = yn;
    }
    if (tid == 0) tL[0] = t_new;
}

__global__ void k_final(const float* __restrict__ z, const float* __restrict__ yhat,
                        float* __restrict__ out, int n) {
    int i = blockIdx.x * blockDim.x + threadIdx.x;
    if (i < n) { out[i] = z[i]; out[n + i] = yhat[i]; }
}

// ---------------------------------------------------------------------------
extern "C" void kernel_launch(void* const* d_in, const int* in_sizes, int n_in,
                              void* d_out, int out_size, void* d_ws, size_t ws_size,
                              hipStream_t stream) {
    if (n_in < 5) return;
    const float* X     = (const float*)d_in[0];
    const float* Y     = (const float*)d_in[1];
    const float* W     = (const float*)d_in[2];
    const float* b     = (const float*)d_in[3];
    const float* gamma = (const float*)d_in[4];

    int n_y   = in_sizes[3];
    int n_x   = in_sizes[2] / n_y;
    int n_obs = in_sizes[1] / n_y;
    int rows  = in_sizes[0] / n_x;   // n_obs + 1
    (void)out_size; (void)ws_size;

    size_t nX = (size_t)rows * n_x;
    size_t nW = (size_t)n_y * n_x;
    size_t nE = (size_t)n_y * n_obs;

    float* ws = (float*)d_ws;
    size_t off = 0;
    auto take  = [&](size_t nflt) { float* p = ws + off; off = (off + nflt + 3) & ~(size_t)3; return p; };
    auto takeb = [&](size_t nbf)  { __bf16* p = (__bf16*)(ws + off); off = (off + (nbf + 1) / 2 + 3) & ~(size_t)3; return p; };

    float*  epT  = take(nE);
    float*  Cov  = take((size_t)n_y * n_y);
    __bf16* Xh   = takeb(nX);
    __bf16* Xl   = takeb(nX);
    __bf16* Wh   = takeb(nW);
    __bf16* Wl   = takeb(nW);
    __bf16* Eh   = takeb(nE);
    __bf16* El   = takeb(nE);
    float*  mu   = take(n_y);
    float*  yhat = take(n_y);
    float*  lin  = take(n_y);
    float*  vp   = take(n_y);
    float*  wv   = take(n_y);
    float*  zv   = take(n_y);
    float*  yv   = take(n_y);
    float*  tL   = take(2);

    // Split inputs to bf16 hi/lo once (streaming, HBM-bound).
    k_split<<<(int)((nX / 4 + 255) / 256), 256, 0, stream>>>(X, Xh, Xl, nX);
    k_split<<<(int)((nW / 4 + 255) / 256), 256, 0, stream>>>(W, Wh, Wl, nW);

    // GEMM1: residuals (transposed) + yhat
    int tiles_m1 = (rows + 15) / 16;
    int tiles_n  = (n_y + 15) / 16;
    int tilesM2a = (tiles_m1 + 1) / 2;
    int tilesN4  = (tiles_n + 3) / 4;
    long waves1  = (long)tilesM2a * tilesN4;
    k_gemm1<<<(int)((waves1 + 7) / 8), 256, 0, stream>>>(
        Xh, Xl, Wh, Wl, b, Y, epT, yhat, rows, n_x, n_y, n_obs, tilesM2a, tilesN4);

    // Column means of ep; split epT to bf16 hi/lo
    k_rowmean<<<(n_y * 32 + 255) / 256, 256, 0, stream>>>(epT, mu, n_obs, n_y);
    k_split<<<(int)((nE / 4 + 255) / 256), 256, 0, stream>>>(epT, Eh, El, nE);

    // GEMM2: Cov
    int tilesM2b = (tiles_n + 1) / 2;
    long waves2  = (long)tilesM2b * tilesN4;
    k_gemm2<<<(int)((waves2 + 7) / 8), 256, 0, stream>>>(
        Eh, El, mu, Cov, n_y, n_obs, tilesM2b, tilesN4);

    // Solver init
    k_init<<<(n_y + 255) / 256, 256, 0, stream>>>(gamma, yhat, lin, zv, yv, vp, tL, n_y);

    int mvb = (n_y * 32 + 255) / 256;
    // Power iteration (30) for Lipschitz constant
    for (int it = 0; it < 30; ++it) {
        k_matvec<<<mvb, 256, 0, stream>>>(Cov, vp, wv, n_y);
        k_normalize<<<1, 1024, 0, stream>>>(wv, vp, n_y);
    }
    k_matvec<<<mvb, 256, 0, stream>>>(Cov, vp, wv, n_y);
    k_L<<<1, 1024, 0, stream>>>(vp, wv, tL, n_y);

    // APG (400 fixed iterations)
    int np2 = 1; while (np2 < n_y) np2 <<= 1;
    size_t sh = (size_t)3 * np2 * sizeof(float);
    for (int it = 0; it < 400; ++it) {
        k_matvec<<<mvb, 256, 0, stream>>>(Cov, yv, wv, n_y);
        k_apg<<<1, 1024, sh, stream>>>(wv, lin, zv, yv, tL, n_y, np2);
    }

    k_final<<<(n_y + 255) / 256, 256, 0, stream>>>(zv, yhat, (float*)d_out, n_y);
}
